// GKernel_41360535061088
// MI455X (gfx1250) — matrix operand order repaired and checked by reading, not compiled, not run on hardware
//
#include <hip/hip_runtime.h>
#include <math.h>

#define Nn 512
#define E1 2048
#define E2 4096
#define Hh 128
#define Ldim 16
#define Ff 16
#define Mm 8
#define NWRD 16            // 512/32 mask words
#define NACT_MAX 448       // cap on active egonet nodes (expected ~70)
#define TAU_INV 10.0f

typedef __attribute__((ext_vector_type(16))) _Float16 v16h;
typedef __attribute__((ext_vector_type(8)))  _Float16 v8h;
typedef __attribute__((ext_vector_type(8)))  float    v8f;

// ---------------- dynamic-LDS layout for k_ego ----------------
#define AL16(x) (((x) + 15) & ~15)
constexpr int LDS_CBUF = NACT_MAX * Hh * 2;                    // f16 c matrix
constexpr int O_C0   = 0;
constexpr int O_C1   = AL16(O_C0 + LDS_CBUF);
constexpr int O_ESRC = AL16(O_C1 + LDS_CBUF);                  // u16[E2] CSR src
constexpr int O_CNT  = AL16(O_ESRC + E2 * 2);                  // u32[NACT_MAX+1]
constexpr int O_ROW  = AL16(O_CNT + (NACT_MAX + 1) * 4);       // u32[NACT_MAX+1]
constexpr int O_G2L  = AL16(O_ROW + (NACT_MAX + 1) * 4);       // i16[Nn]
constexpr int O_ACT  = AL16(O_G2L + Nn * 2);                   // u16[NACT_MAX]
constexpr int O_MASK = AL16(O_ACT + NACT_MAX * 2);             // u32[NWRD]
constexpr int O_FEAT = AL16(O_MASK + NWRD * 4);                // f32[4*Hh]
constexpr int O_STG  = AL16(O_FEAT + 4 * Hh * 4);              // f32[8 waves][16][Hh]
constexpr int O_MISC = AL16(O_STG + 8 * 16 * Hh * 4);
constexpr int LDS_EGO_BYTES = AL16(O_MISC + 16);               // ~304 KB < 320 KB/WGP
static_assert(LDS_EGO_BYTES <= 320 * 1024, "LDS budget");

// ---------------- tiny utility kernels ----------------
__global__ void k_zero(unsigned* __restrict__ p, int n) {
  int i = blockIdx.x * blockDim.x + threadIdx.x;
  if (i < n) p[i] = 0u;
}

__global__ void k_adj(const int* __restrict__ eidx, unsigned* __restrict__ adj) {
  int k = blockIdx.x * blockDim.x + threadIdx.x;
  if (k >= E1) return;
  int u = eidx[k], v = eidx[k + E1];
  atomicOr(&adj[u * NWRD + (v >> 5)], 1u << (v & 31));
  atomicOr(&adj[v * NWRD + (u >> 5)], 1u << (u & 31));
}

// 2-hop reach bitsets: reach[i] = {i} | N1 | N(N1)
__global__ __launch_bounds__(256) void k_reach(const unsigned* __restrict__ adj,
                                               unsigned* __restrict__ reach) {
  __shared__ unsigned r1[NWRD];
  __shared__ unsigned res[NWRD];
  int i = blockIdx.x, tid = threadIdx.x;
  if (tid < NWRD) { r1[tid] = adj[i * NWRD + tid]; res[tid] = 0u; }
  __syncthreads();
  if (tid == 0) r1[i >> 5] |= (1u << (i & 31));
  __syncthreads();
  unsigned accw[NWRD];
#pragma unroll
  for (int w = 0; w < NWRD; ++w) accw[w] = 0u;
  for (int j = tid; j < Nn; j += 256)
    if ((r1[j >> 5] >> (j & 31)) & 1u) {
#pragma unroll
      for (int w = 0; w < NWRD; ++w) accw[w] |= adj[j * NWRD + w];
    }
#pragma unroll
  for (int w = 0; w < NWRD; ++w) if (accw[w]) atomicOr(&res[w], accw[w]);
  __syncthreads();
  if (tid < NWRD) reach[i * NWRD + tid] = res[tid] | r1[tid];
}

__global__ void k_bucket(const float* __restrict__ x, const float* __restrict__ E0,
                         int* __restrict__ bucket) {
  int n = blockIdx.x * blockDim.x + threadIdx.x;
  if (n >= Nn) return;
  int lab = 0;
  for (int l = 0; l < Ldim; ++l) if (x[n * Ldim + l] > 0.5f) { lab = l; break; }
  int b = 0;
  for (int h = 0; h < Hh; ++h) if (E0[lab * Hh + h] > 0.5f) { b = h; break; }
  bucket[n] = b;
}

// Pre-swizzle Rproj [256][128] f32 into per-lane f16 WMMA B-fragments:
// frag index = (kk*8 + nt)*32 + lane, 16 halves each.
// B layout (16-bit, 32x16): lanes0-15: K={0..7,16..23}, N=lane; lanes16-31: K={8..15,24..31}, N=lane-16.
__global__ void k_repack(const float* __restrict__ Rproj, _Float16* __restrict__ bfrag) {
  int id = blockIdx.x * blockDim.x + threadIdx.x;      // 2048 fragment-lanes
  if (id >= 8 * 8 * 32) return;
  int lane = id & 31;
  int tf   = id >> 5;
  int nt   = tf & 7;
  int kk   = tf >> 3;
  int Ncol = nt * 16 + (lane & 15);
  int Kb   = kk * 32 + ((lane >> 4) << 3);
  _Float16* o = bfrag + (size_t)id * 16;
#pragma unroll
  for (int h = 0; h < 16; ++h) o[h] = (_Float16)Rproj[(Kb + (h & 7) + ((h >> 3) << 4)) * Hh + Ncol];
}

// ---------------- the big per-egonet WL kernel (WMMA) ----------------
__global__ __launch_bounds__(256) void k_ego(const int* __restrict__ eidx,
                                             const unsigned* __restrict__ reach,
                                             const int* __restrict__ bucket,
                                             const _Float16* __restrict__ bfrag,
                                             float* __restrict__ ego_feats) {
  extern __shared__ char smem[];
  _Float16*       c0    = (_Float16*)(smem + O_C0);
  _Float16*       c1    = (_Float16*)(smem + O_C1);
  unsigned short* esrc  = (unsigned short*)(smem + O_ESRC);
  unsigned*       cnt   = (unsigned*)(smem + O_CNT);
  unsigned*       roff  = (unsigned*)(smem + O_ROW);
  short*          g2l   = (short*)(smem + O_G2L);
  unsigned short* act   = (unsigned short*)(smem + O_ACT);
  unsigned*       maskw = (unsigned*)(smem + O_MASK);
  float*          feats = (float*)(smem + O_FEAT);
  float*          stage = (float*)(smem + O_STG);
  unsigned*       misc  = (unsigned*)(smem + O_MISC);

  const int tid  = threadIdx.x;
  const int lane = tid & 31;
  const int wv   = tid >> 5;
  const int root = blockIdx.x;

  for (int i = tid; i < NACT_MAX + 1; i += 256) cnt[i] = 0u;
  for (int i = tid; i < Nn; i += 256) g2l[i] = -1;
  for (int i = tid; i < 4 * Hh; i += 256) feats[i] = 0.f;
  if (tid < NWRD) maskw[tid] = reach[root * NWRD + tid];
  if (tid == 0) misc[0] = 0u;
  __syncthreads();

  // compact active nodes
  for (int n = tid; n < Nn; n += 256)
    if ((maskw[n >> 5] >> (n & 31)) & 1u) {
      unsigned la = atomicAdd(&misc[0], 1u);
      if (la < NACT_MAX) { act[la] = (unsigned short)n; g2l[n] = (short)la; }
    }
  __syncthreads();
  int n_act = (int)misc[0]; if (n_act > NACT_MAX) n_act = NACT_MAX;
  int n_pad = (n_act + 15) & ~15;

  for (int i = tid; i < n_pad * Hh; i += 256) { c0[i] = (_Float16)0; c1[i] = (_Float16)0; }
  __syncthreads();

  // c0 one-hot buckets + iteration-0 histogram
  for (int la = tid; la < n_act; la += 256) {
    int b = bucket[act[la]];
    c0[la * Hh + b] = (_Float16)1;
    atomicAdd(&feats[b], 1.0f);
  }
  // in-edge counts (both endpoints active)
  for (int e = tid; e < E2; e += 256) {
    int u = eidx[e], d = eidx[e ^ E1];   // directed edge list = (e0,e1) ++ (e1,e0)
    int lu = g2l[u], ld = g2l[d];
    if (lu >= 0 && ld >= 0) atomicAdd(&cnt[ld], 1u);
  }
  __syncthreads();
  if (tid == 0) {                        // exclusive scan (n_act <= 448)
    unsigned s = 0;
    for (int r = 0; r < n_act; ++r) { roff[r] = s; s += cnt[r]; }
    roff[n_act] = s;
  }
  __syncthreads();
  for (int i = tid; i < NACT_MAX + 1; i += 256) cnt[i] = 0u;
  __syncthreads();
  for (int e = tid; e < E2; e += 256) {  // CSR fill
    int u = eidx[e], d = eidx[e ^ E1];
    int lu = g2l[u], ld = g2l[d];
    if (lu >= 0 && ld >= 0) {
      unsigned pos = roff[ld] + atomicAdd(&cnt[ld], 1u);
      esrc[pos] = (unsigned short)lu;
    }
  }
  __syncthreads();

  float* stg = stage + wv * 16 * Hh;     // per-wave 16x128 f32 staging
  const int cb = lane * 4;               // 4 columns per lane
  const int ntiles = n_pad >> 4;

  for (int t = 0; t < 3; ++t) {
    _Float16* cur = (t & 1) ? c1 : c0;
    _Float16* nxt = (t & 1) ? c0 : c1;

    for (int mt = wv; mt < ntiles; mt += 8) {   // one 16-row M-tile per wave
      const int base = mt * 16;

      for (int i = lane; i < 16 * Hh; i += 32) stg[i] = 0.f;

      // g rows = sum of neighbor c rows (CSR gather)
      for (int r = 0; r < 16; ++r) {
        int row = base + r;
        if (row >= n_act) break;
        unsigned eb = roff[row], ee = roff[row + 1];
        float a0 = 0.f, a1 = 0.f, a2 = 0.f, a3 = 0.f;
        for (unsigned e = eb; e < ee; ++e) {
          const _Float16* cp = cur + (int)esrc[e] * Hh + cb;
          a0 += (float)cp[0]; a1 += (float)cp[1]; a2 += (float)cp[2]; a3 += (float)cp[3];
        }
        float* gp = stg + r * Hh + cb;
        gp[0] = a0; gp[1] = a1; gp[2] = a2; gp[3] = a3;
      }
      asm volatile("" ::: "memory");

      // sig = c_tile @ Rtop + g_tile @ Rbot : 64 x v_wmma_f32_16x16x32_f16
      v8f acc[8];
#pragma unroll
      for (int nt = 0; nt < 8; ++nt)
#pragma unroll
        for (int j = 0; j < 8; ++j) acc[nt][j] = 0.f;

#pragma unroll
      for (int kk = 0; kk < 8; ++kk) {
        v16h a;
        if (kk < 4) {
          // A layout: lanes0-15 row=lane, K = kk*32 + {0..7,16..23}; lanes16-31: +8
          const _Float16* p = cur + (base + (lane & 15)) * Hh + kk * 32 + ((lane >> 4) << 3);
          v8h lo = *(const v8h*)p;
          v8h hi = *(const v8h*)(p + 16);
          a = __builtin_shufflevector(lo, hi, 0,1,2,3,4,5,6,7,8,9,10,11,12,13,14,15);
        } else {
          const float* p = stg + (lane & 15) * Hh + (kk - 4) * 32 + ((lane >> 4) << 3);
#pragma unroll
          for (int i = 0; i < 8; ++i) { a[i] = (_Float16)p[i]; a[i + 8] = (_Float16)p[16 + i]; }
        }
        // Batch-load all 8 B fragments into independent registers so the
        // compiler can emit one clause of global_load_b128 + a single wait,
        // then stream 8 back-to-back WMMAs (no per-WMMA loadcnt stall).
        v16h bfr[8];
#pragma unroll
        for (int nt = 0; nt < 8; ++nt) {
          const v8h* q = (const v8h*)bfrag + (size_t)((kk * 8 + nt) * 32 + lane) * 2;
          bfr[nt] = __builtin_shufflevector(q[0], q[1],
                                            0,1,2,3,4,5,6,7,8,9,10,11,12,13,14,15);
        }
#pragma unroll
        for (int nt = 0; nt < 8; ++nt) {
          acc[nt] = __builtin_amdgcn_wmma_f32_16x16x32_f16(
              false, a, false, bfr[nt], (short)0, acc[nt], false, false);
        }
      }

      // scatter accumulators (C layout) to staging for row-wise softmax
      const int mb = (lane >> 4) << 3;
      const int cl = lane & 15;
#pragma unroll
      for (int nt = 0; nt < 8; ++nt)
#pragma unroll
        for (int j = 0; j < 8; ++j)
          stg[(mb + j) * Hh + nt * 16 + cl] = acc[nt][j];
      asm volatile("" ::: "memory");

      // softmax(sig/tau) per row, write next c, accumulate histogram
      for (int r = 0; r < 16; ++r) {
        int row = base + r;
        if (row >= n_act) break;
        const float* sp = stg + r * Hh + cb;
        float v0 = sp[0], v1 = sp[1], v2 = sp[2], v3 = sp[3];
        float mx = fmaxf(fmaxf(v0, v1), fmaxf(v2, v3));
        for (int o = 16; o > 0; o >>= 1) mx = fmaxf(mx, __shfl_xor(mx, o, 32));
        float e0 = __expf((v0 - mx) * TAU_INV);
        float e1 = __expf((v1 - mx) * TAU_INV);
        float e2 = __expf((v2 - mx) * TAU_INV);
        float e3 = __expf((v3 - mx) * TAU_INV);
        float s = e0 + e1 + e2 + e3;
        for (int o = 16; o > 0; o >>= 1) s += __shfl_xor(s, o, 32);
        float inv = 1.0f / s;
        float p0 = e0 * inv, p1 = e1 * inv, p2 = e2 * inv, p3 = e3 * inv;
        _Float16* np = nxt + row * Hh + cb;
        np[0] = (_Float16)p0; np[1] = (_Float16)p1;
        np[2] = (_Float16)p2; np[3] = (_Float16)p3;
        float* fb = feats + (t + 1) * Hh + cb;
        atomicAdd(fb + 0, p0); atomicAdd(fb + 1, p1);
        atomicAdd(fb + 2, p2); atomicAdd(fb + 3, p3);
      }
    }
    __syncthreads();
  }

  for (int i = tid; i < 4 * Hh; i += 256)
    ego_feats[(size_t)root * (4 * Hh) + i] = feats[i];
}

// ---------------- filter-graph WL (M=8, tiny; plain VALU) ----------------
__global__ __launch_bounds__(128) void k_filt(const float* __restrict__ P,
                                              const float* __restrict__ X,
                                              const float* __restrict__ E0,
                                              const float* __restrict__ Rproj,
                                              float* __restrict__ filt_feats,
                                              float* __restrict__ filt_norm) {
  int f = blockIdx.x, tid = threadIdx.x;   // 128 threads, tid == h
  __shared__ float A[Mm][Mm];
  __shared__ float maskF[Mm];
  __shared__ int   bkt[Mm];
  __shared__ float cF[Mm][Hh], gF[Mm][Hh], sF[Mm][Hh];
  __shared__ float feats[4 * Hh];
  __shared__ float red[Hh];
  if (tid < Mm * Mm) A[tid >> 3][tid & 7] = P[f * Mm * Mm + tid];
  __syncthreads();
  if (tid == 0) {
    int comp[Mm], nc[Mm];
    for (int m = 0; m < Mm; ++m) comp[m] = m;
    for (int it = 0; it < Mm; ++it) {
      for (int m = 0; m < Mm; ++m) {
        int mn = comp[m];
        for (int m2 = 0; m2 < Mm; ++m2)
          if (A[m][m2] > 0.f && comp[m2] < mn) mn = comp[m2];
        nc[m] = mn;
      }
      for (int m = 0; m < Mm; ++m) comp[m] = nc[m];
    }
    int cc[Mm]; for (int m = 0; m < Mm; ++m) cc[m] = 0;
    for (int m = 0; m < Mm; ++m) cc[comp[m]]++;
    int best = 0;
    for (int m = 1; m < Mm; ++m) if (cc[m] > cc[best]) best = m;
    for (int m = 0; m < Mm; ++m) maskF[m] = (comp[m] == best) ? 1.f : 0.f;
    for (int m = 0; m < Mm; ++m) {
      int lab = 0;
      for (int l = 0; l < Ldim; ++l)
        if (X[(f * Mm + m) * Ldim + l] > 0.5f) { lab = l; break; }
      int b = 0;
      for (int h = 0; h < Hh; ++h)
        if (E0[lab * Hh + h] > 0.5f) { b = h; break; }
      bkt[m] = b;
    }
  }
  __syncthreads();
  {
    float s0 = 0.f;
    for (int m = 0; m < Mm; ++m) {
      float v = (bkt[m] == tid) ? maskF[m] : 0.f;
      cF[m][tid] = v; s0 += v;
    }
    feats[tid] = s0;
    for (int i = 1; i < 4; ++i) feats[i * Hh + tid] = 0.f;
  }
  __syncthreads();
  for (int t = 0; t < 3; ++t) {
    for (int m = 0; m < Mm; ++m) {
      float g = 0.f;
      for (int m2 = 0; m2 < Mm; ++m2)
        g += A[m][m2] * maskF[m] * maskF[m2] * cF[m2][tid];
      gF[m][tid] = g;
    }
    __syncthreads();
    for (int m = 0; m < Mm; ++m) {
      float s = 0.f;
      for (int k = 0; k < Hh; ++k)
        s += cF[m][k] * Rproj[k * Hh + tid] + gF[m][k] * Rproj[(k + Hh) * Hh + tid];
      sF[m][tid] = s;
    }
    __syncthreads();
    for (int m = 0; m < Mm; ++m) {
      red[tid] = sF[m][tid];
      __syncthreads();
      for (int o = 64; o > 0; o >>= 1) { if (tid < o) red[tid] = fmaxf(red[tid], red[tid + o]); __syncthreads(); }
      float mx = red[0];
      __syncthreads();
      float ev = __expf((sF[m][tid] - mx) * TAU_INV);
      red[tid] = ev;
      __syncthreads();
      for (int o = 64; o > 0; o >>= 1) { if (tid < o) red[tid] += red[tid + o]; __syncthreads(); }
      float nv = maskF[m] * ev / red[0];
      __syncthreads();
      cF[m][tid] = nv;
      feats[(t + 1) * Hh + tid] += nv;
      __syncthreads();
    }
  }
  float p2 = 0.f;
  for (int i = tid; i < 4 * Hh; i += 128) {
    float v = feats[i];
    filt_feats[f * 4 * Hh + i] = v;
    p2 += v * v;
  }
  red[tid] = p2;
  __syncthreads();
  for (int o = 64; o > 0; o >>= 1) { if (tid < o) red[tid] += red[tid + o]; __syncthreads(); }
  if (tid == 0) filt_norm[f] = sqrtf(red[0]);
}

// ---------------- normalized similarity ----------------
__global__ __launch_bounds__(128) void k_sim(const float* __restrict__ ego_feats,
                                             const float* __restrict__ filt_feats,
                                             const float* __restrict__ filt_norm,
                                             float* __restrict__ out) {
  int n = blockIdx.x, tid = threadIdx.x;
  __shared__ float ego[4 * Hh];
  __shared__ float red[128];
  __shared__ float ngs;
  for (int i = tid; i < 4 * Hh; i += 128) ego[i] = ego_feats[(size_t)n * 4 * Hh + i];
  __syncthreads();
  float p = 0.f;
  for (int i = tid; i < 4 * Hh; i += 128) p += ego[i] * ego[i];
  red[tid] = p;
  __syncthreads();
  for (int o = 64; o > 0; o >>= 1) { if (tid < o) red[tid] += red[tid + o]; __syncthreads(); }
  if (tid == 0) ngs = sqrtf(red[0]);
  __syncthreads();
  for (int f = 0; f < Ff; ++f) {
    float d = 0.f;
    for (int i = tid; i < 4 * Hh; i += 128) d += ego[i] * filt_feats[f * 4 * Hh + i];
    red[tid] = d;
    __syncthreads();
    for (int o = 64; o > 0; o >>= 1) { if (tid < o) red[tid] += red[tid + o]; __syncthreads(); }
    if (tid == 0) out[n * Ff + f] = red[0] / (filt_norm[f] * ngs + 1e-12f);
    __syncthreads();
  }
}

// ---------------- launch ----------------
extern "C" void kernel_launch(void* const* d_in, const int* in_sizes, int n_in,
                              void* d_out, int out_size, void* d_ws, size_t ws_size,
                              hipStream_t stream) {
  const float* x     = (const float*)d_in[0];
  const int*   eidx  = (const int*)d_in[1];
  const float* P     = (const float*)d_in[2];
  const float* X     = (const float*)d_in[3];
  const float* E0    = (const float*)d_in[4];
  const float* Rproj = (const float*)d_in[5];
  float* out = (float*)d_out;

  char* ws = (char*)d_ws;
  size_t off = 0;
  auto take = [&](size_t bytes) {
    char* p = ws + off;
    off = (off + bytes + 255) & ~(size_t)255;
    return p;
  };
  unsigned* adj        = (unsigned*)take((size_t)Nn * NWRD * 4);
  unsigned* reach      = (unsigned*)take((size_t)Nn * NWRD * 4);
  int*      bucket     = (int*)take((size_t)Nn * 4);
  _Float16* bfrag      = (_Float16*)take((size_t)8 * 8 * 32 * 16 * 2);
  float*    ego_feats  = (float*)take((size_t)Nn * 4 * Hh * 4);
  float*    filt_feats = (float*)take((size_t)Ff * 4 * Hh * 4);
  float*    filt_norm  = (float*)take((size_t)Ff * 4);
  (void)in_sizes; (void)n_in; (void)out_size; (void)ws_size;

  k_zero  <<<(Nn * NWRD + 255) / 256, 256, 0, stream>>>(adj, Nn * NWRD);
  k_adj   <<<(E1 + 255) / 256, 256, 0, stream>>>(eidx, adj);
  k_reach <<<Nn, 256, 0, stream>>>(adj, reach);
  k_bucket<<<(Nn + 255) / 256, 256, 0, stream>>>(x, E0, bucket);
  k_repack<<<8, 256, 0, stream>>>(Rproj, bfrag);
  k_ego   <<<Nn, 256, LDS_EGO_BYTES, stream>>>(eidx, reach, bucket, bfrag, ego_feats);
  k_filt  <<<Ff, 128, 0, stream>>>(P, X, E0, Rproj, filt_feats, filt_norm);
  k_sim   <<<Nn, 128, 0, stream>>>(ego_feats, filt_feats, filt_norm, out);
}
